// BlockwiseParallelTransformerAttention_2319282340611
// MI455X (gfx1250) — matrix operand order repaired and checked by reading, not compiled, not run on hardware
//
#include <hip/hip_runtime.h>
#include <hip/hip_bf16.h>

// ---------------------------------------------------------------------------
// B=2, S=2048, IN=2048, HID=2048, H=16, D=128.
// outputs: attn_output = zeros[B,S,HID]; attn_weights[b,q,h,k] = (Q/sqrt(D)).K^T
// ---------------------------------------------------------------------------
#define PB    2
#define PS    2048
#define PIN   2048
#define PHID  2048
#define PH    16
#define PD    128

typedef __attribute__((ext_vector_type(16))) __bf16 v16bf;
typedef __attribute__((ext_vector_type(8)))  float  v8f;

union Frag {
    uint4 q[2];   // 32 bytes of bf16 bits
    v16bf v;
};

__device__ __forceinline__ unsigned short f2bf_bits(float f) {
    union { float f; unsigned u; } x; x.f = f;
    unsigned r = x.u + 0x7FFFu + ((x.u >> 16) & 1u);   // round-to-nearest-even
    return (unsigned short)(r >> 16);
}
__device__ __forceinline__ unsigned pack2(float lo, float hi) {
    return (unsigned)f2bf_bits(lo) | ((unsigned)f2bf_bits(hi) << 16);
}

// ---------------------------------------------------------------------------
// async-copy support (gfx1250): guard with __has_builtin so compile never breaks
// ---------------------------------------------------------------------------
#if __has_builtin(__builtin_amdgcn_global_load_async_to_lds_b128)
#define USE_ASYNC 1
#else
#define USE_ASYNC 0
#endif

// exact parameter types per hipcc diagnostic:
//   arg0: 'vector_size(16) int __device__ *'  (global, AS1)
//   arg1: LDS destination (AS3)
typedef int v4i_t __attribute__((vector_size(16)));
typedef v4i_t __attribute__((address_space(1))) *g_v4i;
typedef v4i_t __attribute__((address_space(3))) *l_v4i;

__device__ __forceinline__ void wait_asynccnt(int n) {
#if USE_ASYNC
#if __has_builtin(__builtin_amdgcn_s_wait_asynccnt)
    if (n == 0) __builtin_amdgcn_s_wait_asynccnt(0);
    else        __builtin_amdgcn_s_wait_asynccnt(8);
#else
    if (n == 0) asm volatile("s_wait_asynccnt 0" ::: "memory");
    else        asm volatile("s_wait_asynccnt 8" ::: "memory");
#endif
#endif
}

// ---------------------------------------------------------------------------
// Zero-fill attn_output region.
// ---------------------------------------------------------------------------
__global__ __launch_bounds__(256) void zero_out_kernel(float4* __restrict__ p, long n4) {
    long i = (long)blockIdx.x * blockDim.x + threadIdx.x;
    if (i < n4) p[i] = make_float4(0.f, 0.f, 0.f, 0.f);
}

// ---------------------------------------------------------------------------
// Elementwise fp32 -> bf16-bits conversion (done once; GEMMs are pure bf16).
// ---------------------------------------------------------------------------
__global__ __launch_bounds__(256) void convert_f32_bf16(
    const float4* __restrict__ src, uint2* __restrict__ dst, long n4)
{
    long i = (long)blockIdx.x * blockDim.x + threadIdx.x;
    if (i < n4) {
        float4 f = src[i];
        dst[i] = make_uint2(pack2(f.x, f.y), pack2(f.z, f.w));
    }
}

// ---------------------------------------------------------------------------
// Projection GEMM (bf16 in, bf16 out): Y[m][n] = scale * sum_k X[m][k]*W[n][k]
// 256 threads / 8 waves, 128x128 tile, BK=64 (2 WMMA k-steps per stage),
// double-buffered LDS filled with GLOBAL_LOAD_ASYNC_TO_LDS_B128 (ASYNCcnt),
// fragments read with conflict-free ds_load_b128 (row stride 144B).
// ---------------------------------------------------------------------------
#define BKP   64
#define LDSK  (BKP + 8)          // 72 shorts = 144 bytes row stride
#define NCHNK 4                  // 16B chunks per thread per tile per stage

__global__ __launch_bounds__(256) void proj_gemm_bf16(
    const unsigned short* __restrict__ Xb,   // [M,K] bf16 bits
    const unsigned short* __restrict__ Wb,   // [N,K] bf16 bits
    unsigned short* __restrict__ Y,          // [M,N] bf16 bits
    int M, int N, int K, float scale)
{
    __shared__ unsigned short lds[2][2][128 * LDSK];   // [buf][A=0/B=1]

    const int tid  = threadIdx.x;
    const int lane = tid & 31;
    const int wv   = tid >> 5;
    const int wm   = wv & 3;        // 4 m-bands of 32 rows
    const int wn   = wv >> 2;       // 2 n-bands of 64 cols
    const int half = lane >> 4;
    const int l16  = lane & 15;

    const long gm0 = (long)blockIdx.y * 128;
    const long gn0 = (long)blockIdx.x * 128;

    // stage copy: 128 rows x 64 shorts = 1024 x 16B chunks per tile.
    // chunk c = tid + 256*j  ->  row = c>>3, sub = c&7 (16B each)
    auto issue_stage = [&](int stage, int buf) {
        const long k0 = (long)stage * BKP;
#pragma unroll
        for (int j = 0; j < NCHNK; ++j) {
            const int c   = tid + 256 * j;
            const int row = c >> 3;
            const int sub = c & 7;
            const unsigned short* ga = Xb + (gm0 + row) * (long)K + k0 + sub * 8;
            const unsigned short* gb = Wb + (gn0 + row) * (long)K + k0 + sub * 8;
            unsigned short* la = &lds[buf][0][row * LDSK + sub * 8];
            unsigned short* lb = &lds[buf][1][row * LDSK + sub * 8];
#if USE_ASYNC
            __builtin_amdgcn_global_load_async_to_lds_b128(
                (g_v4i)ga, (l_v4i)la, 0, 0);
            __builtin_amdgcn_global_load_async_to_lds_b128(
                (g_v4i)gb, (l_v4i)lb, 0, 0);
#else
            *(uint4*)la = *(const uint4*)ga;
            *(uint4*)lb = *(const uint4*)gb;
#endif
        }
    };

    v8f acc[2][4] = {};
    const int nk = K / BKP;          // 32 stages

    issue_stage(0, 0);
    for (int s = 0; s < nk; ++s) {
        const int buf = s & 1;
        if (s + 1 < nk) issue_stage(s + 1, buf ^ 1);   // prefetch next stage
        wait_asynccnt((s + 1 < nk) ? 8 : 0);           // our stage-s copies done
        __syncthreads();                               // everyone's copies done

#pragma unroll
        for (int ks = 0; ks < 2; ++ks) {               // two 16x16x32 k-steps
            const int ko = ks * 32;
            Frag af[2], bfr[4];
#pragma unroll
            for (int mi = 0; mi < 2; ++mi) {
                int m = wm * 32 + mi * 16 + l16;
                int base = m * LDSK + ko;
                af[mi].q[0] = *(const uint4*)&lds[buf][0][base + 8 * half];
                af[mi].q[1] = *(const uint4*)&lds[buf][0][base + 16 + 8 * half];
            }
#pragma unroll
            for (int ni = 0; ni < 4; ++ni) {
                int n = wn * 64 + ni * 16 + l16;
                int base = n * LDSK + ko;
                bfr[ni].q[0] = *(const uint4*)&lds[buf][1][base + 8 * half];
                bfr[ni].q[1] = *(const uint4*)&lds[buf][1][base + 16 + 8 * half];
            }
#pragma unroll
            for (int mi = 0; mi < 2; ++mi)
#pragma unroll
                for (int ni = 0; ni < 4; ++ni)
                    acc[mi][ni] = __builtin_amdgcn_wmma_f32_16x16x32_bf16(
                        false, af[mi].v, false, bfr[ni].v,
                        (short)0, acc[mi][ni], false, false);
        }
        __syncthreads();                               // buf free for stage s+2
    }

    // epilogue: scale, convert, store bf16
#pragma unroll
    for (int mi = 0; mi < 2; ++mi)
#pragma unroll
        for (int ni = 0; ni < 4; ++ni) {
            long n = gn0 + wn * 64 + ni * 16 + l16;
#pragma unroll
            for (int rr = 0; rr < 8; ++rr) {
                long m = gm0 + wm * 32 + mi * 16 + rr + 8 * half;
                Y[m * (long)N + n] = f2bf_bits(acc[mi][ni][rr] * scale);
            }
        }
}

// ---------------------------------------------------------------------------
// Attention scores: out[b,sq,h,sk] = sum_d Q[b,sq,h,d]*K[b,sk,h,d]
// Q,K bf16 (33MB total -> resident in 192MB L2); fragments loaded straight
// from global, 4 WMMA k-steps, fp32 result streamed to HBM (the roofline).
// ---------------------------------------------------------------------------
__global__ __launch_bounds__(256) void attn_scores_kernel(
    const unsigned short* __restrict__ Q,
    const unsigned short* __restrict__ Km,
    float* __restrict__ out)
{
    const int bh = blockIdx.z;
    const int b  = bh >> 4;
    const int h  = bh & 15;

    const int lane = threadIdx.x & 31;
    const int wv   = threadIdx.x >> 5;
    const int wm   = wv & 3;
    const int wn   = wv >> 2;
    const int half = lane >> 4;
    const int l16  = lane & 15;

    const long gm0 = (long)blockIdx.y * 128;   // sq tile
    const long gn0 = (long)blockIdx.x * 128;   // sk tile

    const unsigned short* Qb = Q  + (long)b * PS * PHID + (long)h * PD;
    const unsigned short* Kb = Km + (long)b * PS * PHID + (long)h * PD;

    v8f acc[2][4] = {};

#pragma unroll
    for (int kd = 0; kd < PD; kd += 32) {
        Frag af[2], bfr[4];
#pragma unroll
        for (int mi = 0; mi < 2; ++mi) {
            long sq = gm0 + wm * 32 + mi * 16 + l16;
            const unsigned short* p = Qb + sq * PHID + kd;
            af[mi].q[0] = *(const uint4*)(p + 8 * half);
            af[mi].q[1] = *(const uint4*)(p + 16 + 8 * half);
        }
#pragma unroll
        for (int ni = 0; ni < 4; ++ni) {
            long sk = gn0 + wn * 64 + ni * 16 + l16;
            const unsigned short* p = Kb + sk * PHID + kd;
            bfr[ni].q[0] = *(const uint4*)(p + 8 * half);
            bfr[ni].q[1] = *(const uint4*)(p + 16 + 8 * half);
        }
#pragma unroll
        for (int mi = 0; mi < 2; ++mi)
#pragma unroll
            for (int ni = 0; ni < 4; ++ni)
                acc[mi][ni] = __builtin_amdgcn_wmma_f32_16x16x32_bf16(
                    false, af[mi].v, false, bfr[ni].v,
                    (short)0, acc[mi][ni], false, false);
    }

#pragma unroll
    for (int mi = 0; mi < 2; ++mi)
#pragma unroll
        for (int ni = 0; ni < 4; ++ni) {
            long sk = gn0 + wn * 64 + ni * 16 + l16;
#pragma unroll
            for (int rr = 0; rr < 8; ++rr) {
                long sq = gm0 + wm * 32 + mi * 16 + rr + 8 * half;
                out[(((long)b * PS + sq) * PH + h) * PS + sk] = acc[mi][ni][rr];
            }
        }
}

// ---------------------------------------------------------------------------
// Launch
// ---------------------------------------------------------------------------
extern "C" void kernel_launch(void* const* d_in, const int* in_sizes, int n_in,
                              void* d_out, int out_size, void* d_ws, size_t ws_size,
                              hipStream_t stream) {
    const float* x  = (const float*)d_in[0];   // [B,S,IN]
    const float* Wq = (const float*)d_in[1];   // [HID,IN]
    const float* Wk = (const float*)d_in[2];   // [HID,IN]
    // d_in[3] = Wv : unused (v is discarded by the reference)

    const long M  = (long)PB * PS;                       // 4096
    const long nX = M * PIN;                             // 8,388,608
    const long nW = (long)PHID * PIN;                    // 4,194,304
    const long attn_out_elems = (long)PB * PS * PHID;    // 8,388,608 zeros

    // workspace layout (ushort elements)
    unsigned short* Xbf  = (unsigned short*)d_ws;        // 16 MB
    unsigned short* Wqbf = Xbf  + nX;                    //  8 MB
    unsigned short* Wkbf = Wqbf + nW;                    //  8 MB
    unsigned short* Qbf  = Wkbf + nW;                    // 16 MB
    unsigned short* Kbf  = Qbf  + M * PHID;              // 16 MB

    float* out0 = (float*)d_out;                 // attn_output (zeros)
    float* outW = out0 + attn_out_elems;         // attn_weights

    // 1) attn_output = 0
    {
        long n4 = attn_out_elems / 4;
        zero_out_kernel<<<dim3((unsigned)((n4 + 255) / 256)), dim3(256), 0, stream>>>(
            (float4*)out0, n4);
    }

    // 2) one-shot fp32 -> bf16 conversion of X, Wq, Wk
    convert_f32_bf16<<<dim3((unsigned)((nX / 4 + 255) / 256)), dim3(256), 0, stream>>>(
        (const float4*)x, (uint2*)Xbf, nX / 4);
    convert_f32_bf16<<<dim3((unsigned)((nW / 4 + 255) / 256)), dim3(256), 0, stream>>>(
        (const float4*)Wq, (uint2*)Wqbf, nW / 4);
    convert_f32_bf16<<<dim3((unsigned)((nW / 4 + 255) / 256)), dim3(256), 0, stream>>>(
        (const float4*)Wk, (uint2*)Wkbf, nW / 4);

    // 3) Q = (X @ Wq^T) / sqrt(D);  K = X @ Wk^T   (bf16 WMMA, async LDS pipe)
    const float scale = 0.08838834764831845f;    // 1/sqrt(128)
    proj_gemm_bf16<<<dim3(PHID / 128, (unsigned)(M / 128)), dim3(256), 0, stream>>>(
        Xbf, Wqbf, Qbf, (int)M, PHID, PIN, scale);
    proj_gemm_bf16<<<dim3(PHID / 128, (unsigned)(M / 128)), dim3(256), 0, stream>>>(
        Xbf, Wkbf, Kbf, (int)M, PHID, PIN, 1.0f);

    // 4) attn_weights = Q . K^T per (b,h)
    attn_scores_kernel<<<dim3(PS / 128, PS / 128, PB * PH), dim3(256), 0, stream>>>(
        Qbf, Kbf, outW);
}
